// Weighted_Dist_UDF_10376640987891
// MI455X (gfx1250) — compile-verified, matrix operand order checked by reading, not run on hardware
//
#include <hip/hip_runtime.h>
#include <math.h>

// ---------------------------------------------------------------------------
// Problem constants (from reference)
// ---------------------------------------------------------------------------
#define BATCH 2
#define NP    4096
#define NQ    8192
#define KNN   10
#define MROWS (BATCH * NQ * KNN)     // 163840 rows through the MLPs
#define MTILES (MROWS / 16)          // 10240
#define BN_EPS 1e-5f
#define LRELU 0.2f

typedef __bf16 bf16_t;
typedef __attribute__((ext_vector_type(16))) __bf16 v16bf;
typedef __attribute__((ext_vector_type(8)))  __bf16 v8bf;
typedef __attribute__((ext_vector_type(8)))  float  v8f;

// ---------------------------------------------------------------------------
// 1) Weight prep: fold BN scale into W, convert to bf16, pre-swizzle into the
//    exact WMMA B-fragment layout: [ntile][kchunk][lane][16 elems].
//    lane<16  : col n = ntile*16+lane,     K elems {0..7, 16..23} of chunk
//    lane>=16 : col n = ntile*16+lane-16,  K elems {8..15, 24..31} of chunk
// ---------------------------------------------------------------------------
__global__ void pack_weights(const float* __restrict__ w, const float* __restrict__ b,
                             const float* __restrict__ g, const float* __restrict__ be,
                             const float* __restrict__ mu, const float* __restrict__ va,
                             int Ci, int Co, int Kchunks, int has_bn,
                             bf16_t* __restrict__ wout, float* __restrict__ bout)
{
    int tid   = blockIdx.x * blockDim.x + threadIdx.x;
    int total = (Co / 16) * Kchunks * 512;

    if (tid < Co) {
        float bias = b[tid];
        if (has_bn) {
            float s = g[tid] * rsqrtf(va[tid] + BN_EPS);
            bias = s * (bias - mu[tid]) + be[tid];
        }
        bout[tid] = bias;
    }
    if (tid >= total) return;

    int e      = tid & 15;
    int lane   = (tid >> 4) & 31;
    int chunk  = tid >> 9;              // ntile*Kchunks + kchunk
    int kchunk = chunk % Kchunks;
    int ntile  = chunk / Kchunks;

    int n      = ntile * 16 + (lane & 15);
    int klocal = (lane < 16) ? ((e < 8) ? e : e + 8)
                             : ((e < 8) ? e + 8 : e + 16);
    int k      = kchunk * 32 + klocal;

    float v = 0.0f;
    if (n < Co && k < Ci) {
        v = w[n * Ci + k];
        if (has_bn) v *= g[n] * rsqrtf(va[n] + BN_EPS);
    }
    wout[tid] = (bf16_t)v;
}

// ---------------------------------------------------------------------------
// 2) Brute-force KNN. Point cloud for the batch lives in LDS (48 KB).
//    One thread per query; register-resident top-10 (unordered — downstream
//    ops are symmetric in k, so no sort needed). Emits the padded bf16 input
//    activation row [local(3), q(3), 0...] plus local (f32) and kdist (f32).
// ---------------------------------------------------------------------------
__global__ __launch_bounds__(256) void knn_kernel(
    const float* __restrict__ pcd,      // (B, NP, 3)
    const float* __restrict__ qp,       // (B, 3, NQ)
    bf16_t* __restrict__ act0,          // (MROWS, 32)
    float*  __restrict__ localbuf,      // (MROWS, 3)
    float*  __restrict__ kdistbuf)      // (MROWS)
{
    __shared__ float spcd[NP * 3];      // 48 KB
    const int b  = blockIdx.y;
    const int qi = blockIdx.x * 256 + threadIdx.x;

    for (int i = threadIdx.x; i < NP * 3; i += 256)
        spcd[i] = pcd[(size_t)b * NP * 3 + i];
    __syncthreads();

    const float qx = qp[(size_t)b * 3 * NQ + 0 * NQ + qi];
    const float qy = qp[(size_t)b * 3 * NQ + 1 * NQ + qi];
    const float qz = qp[(size_t)b * 3 * NQ + 2 * NQ + qi];

    float bd[KNN];
    int   bi[KNN];
#pragma unroll
    for (int j = 0; j < KNN; ++j) { bd[j] = 3.0e38f; bi[j] = 0; }
    float wv = 3.0e38f;
    int   wp = 0;

    for (int p = 0; p < NP; ++p) {
        const float dx = qx - spcd[3 * p + 0];
        const float dy = qy - spcd[3 * p + 1];
        const float dz = qz - spcd[3 * p + 2];
        const float d2 = dx * dx + dy * dy + dz * dz;
        if (d2 < wv) {
            // predicated (register-resident) replace of current worst
#pragma unroll
            for (int j = 0; j < KNN; ++j)
                if (j == wp) { bd[j] = d2; bi[j] = p; }
            // rescan for new worst
            wv = bd[0]; wp = 0;
#pragma unroll
            for (int j = 1; j < KNN; ++j)
                if (bd[j] > wv) { wv = bd[j]; wp = j; }
        }
    }

#pragma unroll
    for (int k = 0; k < KNN; ++k) {
        const int p = bi[k];
        const float lx = qx - spcd[3 * p + 0];
        const float ly = qy - spcd[3 * p + 1];
        const float lz = qz - spcd[3 * p + 2];
        const size_t row = ((size_t)(b * NQ + qi)) * KNN + k;
        bf16_t* a = act0 + row * 32;
        a[0] = (bf16_t)lx; a[1] = (bf16_t)ly; a[2] = (bf16_t)lz;
        a[3] = (bf16_t)qx; a[4] = (bf16_t)qy; a[5] = (bf16_t)qz;
#pragma unroll
        for (int j = 6; j < 32; ++j) a[j] = (bf16_t)0.0f;
        localbuf[row * 3 + 0] = lx;
        localbuf[row * 3 + 1] = ly;
        localbuf[row * 3 + 2] = lz;
        kdistbuf[row] = sqrtf(lx * lx + ly * ly + lz * lz);
    }
}

// ---------------------------------------------------------------------------
// 3) Register-blocked GEMM layer: out = act(W * in + b), bf16 in/out,
//    f32 accumulate via v_wmma_f32_16x16x32_bf16.
//    256 threads = 8 waves; each wave owns a 2-mtile x NB-ntile block of
//    16x16 D tiles. Per 32-wide K chunk: 2 A-fragment loads + NB B-fragment
//    loads feed 2*NB WMMAs (4x the math per activation byte vs 1x1 tiling;
//    activation re-reads drop from N/16 to N/(16*NB)).
// ---------------------------------------------------------------------------
template<int NB>
__global__ __launch_bounds__(256) void gemm_bf16(
    const bf16_t* __restrict__ in, int ldin,
    const bf16_t* __restrict__ wp,
    const float*  __restrict__ bias,
    bf16_t* __restrict__ out, int ldout,
    int kchunks, float slope, int act)
{
    const int lane   = threadIdx.x & 31;
    const int wave   = threadIdx.x >> 5;
    const int mtile0 = (blockIdx.y * 8 + wave) * 2;   // MB = 2
    const int ntile0 = blockIdx.x * NB;

    // A fragment sources: row = mtile*16 + (lane&15); +8 K-offset for hi lanes
    const bf16_t* arow0 = in + (size_t)(mtile0 * 16 + (lane & 15)) * ldin
                             + ((lane >> 4) << 3);
    const bf16_t* arow1 = arow0 + (size_t)16 * ldin;
    // B fragment base: pre-swizzled, 16 contiguous bf16 per lane per chunk;
    // ntile stride = kchunks*512 elems, kchunk stride = 512 elems.
    const bf16_t* bbase = wp + ((size_t)ntile0 * kchunks * 32 + lane) * 16;

    v8f c[2][NB];
    const v8f zero = {0.f, 0.f, 0.f, 0.f, 0.f, 0.f, 0.f, 0.f};
#pragma unroll
    for (int mb = 0; mb < 2; ++mb)
#pragma unroll
        for (int nb = 0; nb < NB; ++nb)
            c[mb][nb] = zero;

    for (int kc = 0; kc < kchunks; ++kc) {
        union { v16bf v; v8bf h[2]; } a0, a1;
        a0.h[0] = *(const v8bf*)(arow0);        // K {0..7} / {8..15}
        a0.h[1] = *(const v8bf*)(arow0 + 16);   // K {16..23} / {24..31}
        a1.h[0] = *(const v8bf*)(arow1);
        a1.h[1] = *(const v8bf*)(arow1 + 16);
#pragma unroll
        for (int nb = 0; nb < NB; ++nb) {
            v16bf bm = *(const v16bf*)(bbase + (size_t)nb * kchunks * 512
                                             + (size_t)kc * 512);
            c[0][nb] = __builtin_amdgcn_wmma_f32_16x16x32_bf16(
                           false, a0.v, false, bm, (short)0, c[0][nb], false, false);
            c[1][nb] = __builtin_amdgcn_wmma_f32_16x16x32_bf16(
                           false, a1.v, false, bm, (short)0, c[1][nb], false, false);
        }
        arow0 += 32;
        arow1 += 32;
    }

#pragma unroll
    for (int mb = 0; mb < 2; ++mb) {
        const int mbase = (mtile0 + mb) * 16 + ((lane >> 4) << 3);
#pragma unroll
        for (int nb = 0; nb < NB; ++nb) {
            const int col  = (ntile0 + nb) * 16 + (lane & 15);
            const float bb = bias[col];
#pragma unroll
            for (int i = 0; i < 8; ++i) {
                float v = c[mb][nb][i] + bb;
                if (act) v = (v >= 0.0f) ? v : slope * v;
                out[(size_t)(mbase + i) * ldout + col] = (bf16_t)v;
            }
        }
    }
}

// ---------------------------------------------------------------------------
// 4) Build cv2 (288-wide, padded from 263): [local3,q3,kdist, feat128, pf128]
//    pf = max over the K=10 rows of each (b,q) group.
// ---------------------------------------------------------------------------
__global__ __launch_bounds__(160) void concat_maxpool(
    const bf16_t* __restrict__ feat,    // (MROWS, 128)
    const bf16_t* __restrict__ act0,    // (MROWS, 32)  cols 0..5 = cv1
    const float*  __restrict__ kdist,   // (MROWS)
    bf16_t* __restrict__ cv2)           // (MROWS, 288)
{
    const size_t row0 = (size_t)blockIdx.x * KNN;
    const int c = threadIdx.x;

    if (c < 128) {
        float mx = -3.0e38f;
#pragma unroll
        for (int k = 0; k < KNN; ++k) {
            const size_t r = row0 + k;
            const float f = (float)feat[r * 128 + c];
            cv2[r * 288 + 7 + c] = (bf16_t)f;
            mx = fmaxf(mx, f);
        }
#pragma unroll
        for (int k = 0; k < KNN; ++k)
            cv2[(row0 + k) * 288 + 135 + c] = (bf16_t)mx;
    } else if (c < 134) {
        const int cc = c - 128;
#pragma unroll
        for (int k = 0; k < KNN; ++k)
            cv2[(row0 + k) * 288 + cc] = act0[(row0 + k) * 32 + cc];
    } else if (c == 134) {
#pragma unroll
        for (int k = 0; k < KNN; ++k)
            cv2[(row0 + k) * 288 + 6] = (bf16_t)kdist[row0 + k];
    } else { // zero-pad cols 263..287
        const int col = 263 + (c - 135);
#pragma unroll
        for (int k = 0; k < KNN; ++k)
            cv2[(row0 + k) * 288 + col] = (bf16_t)0.0f;
    }
}

// ---------------------------------------------------------------------------
// 5) Final attn layer 32 -> 1 (no BN / no act): plain dot product per row.
// ---------------------------------------------------------------------------
__global__ __launch_bounds__(256) void attn_last(
    const bf16_t* __restrict__ in,      // (MROWS, 32)
    const float*  __restrict__ w,       // (1, 32)
    const float*  __restrict__ b,       // (1)
    float* __restrict__ logits)         // (MROWS)
{
    const int m = blockIdx.x * 256 + threadIdx.x;
    if (m >= MROWS) return;
    const bf16_t* row = in + (size_t)m * 32;
    float s = b[0];
#pragma unroll
    for (int j = 0; j < 32; ++j)
        s += (float)row[j] * w[j];
    logits[m] = s;
}

// ---------------------------------------------------------------------------
// 6) Softmax over K, weighted sum of local offsets, L2 norm -> udf.
// ---------------------------------------------------------------------------
__global__ __launch_bounds__(256) void softmax_udf(
    const float* __restrict__ logits,   // (MROWS)
    const float* __restrict__ localbuf, // (MROWS, 3)
    float* __restrict__ out)            // (B*NQ)
{
    const int g = blockIdx.x * 256 + threadIdx.x;
    if (g >= BATCH * NQ) return;
    const size_t row0 = (size_t)g * KNN;

    float l[KNN];
    float mx = -3.0e38f;
#pragma unroll
    for (int k = 0; k < KNN; ++k) { l[k] = logits[row0 + k]; mx = fmaxf(mx, l[k]); }
    float sum = 0.0f;
#pragma unroll
    for (int k = 0; k < KNN; ++k) { l[k] = __expf(l[k] - mx); sum += l[k]; }
    const float inv = 1.0f / sum;

    float ex = 0.f, ey = 0.f, ez = 0.f;
#pragma unroll
    for (int k = 0; k < KNN; ++k) {
        const float wk = l[k] * inv;
        ex += wk * localbuf[(row0 + k) * 3 + 0];
        ey += wk * localbuf[(row0 + k) * 3 + 1];
        ez += wk * localbuf[(row0 + k) * 3 + 2];
    }
    out[g] = sqrtf(ex * ex + ey * ey + ez * ez);
}

// ---------------------------------------------------------------------------
// Host-side orchestration
// ---------------------------------------------------------------------------
static inline size_t alignUp(size_t x, size_t a) { return (x + a - 1) & ~(a - 1); }

extern "C" void kernel_launch(void* const* d_in, const int* in_sizes, int n_in,
                              void* d_out, int out_size, void* d_ws, size_t ws_size,
                              hipStream_t stream)
{
    (void)in_sizes; (void)n_in; (void)out_size; (void)ws_size;

    const float* input_pcd = (const float*)d_in[0];    // (B, NP, 3)
    const float* query_pts = (const float*)d_in[1];    // (B, 3, NQ)

    // Param leaves flattened with JAX pytree ordering (dict keys sorted):
    // top: attn < patch; per convbn layer: b, be, g, mu, va, w; per conv: b, w.
    struct BN { const float *b, *be, *g, *mu, *va, *w; };
    auto cbn = [&](int i0) {
        BN r; r.b  = (const float*)d_in[i0 + 0]; r.be = (const float*)d_in[i0 + 1];
              r.g  = (const float*)d_in[i0 + 2]; r.mu = (const float*)d_in[i0 + 3];
              r.va = (const float*)d_in[i0 + 4]; r.w  = (const float*)d_in[i0 + 5];
        return r;
    };
    BN A0 = cbn(2), A1 = cbn(8), A2 = cbn(14);
    const float* A3b = (const float*)d_in[20];
    const float* A3w = (const float*)d_in[21];
    BN P0 = cbn(22), P1 = cbn(28), P2 = cbn(34);
    const float* P3b = (const float*)d_in[40];
    const float* P3w = (const float*)d_in[41];

    // --- workspace layout -------------------------------------------------
    char* ws = (char*)d_ws;
    size_t off = 0;
    auto take = [&](size_t bytes) { size_t o = off; off = alignUp(off + bytes, 256); return o; };

    // packed weights: elems = (Co/16) * Kchunks * 512
    const size_t wpP0 = take((size_t)( 64/16) * 1 * 512 * 2);
    const size_t wpP1 = take((size_t)(128/16) * 2 * 512 * 2);
    const size_t wpP2 = take((size_t)(128/16) * 4 * 512 * 2);
    const size_t wpP3 = take((size_t)(128/16) * 4 * 512 * 2);
    const size_t wpA0 = take((size_t)(256/16) * 9 * 512 * 2);
    const size_t wpA1 = take((size_t)(128/16) * 8 * 512 * 2);
    const size_t wpA2 = take((size_t)( 32/16) * 4 * 512 * 2);
    const size_t bP0 = take( 64 * 4), bP1 = take(128 * 4), bP2 = take(128 * 4);
    const size_t bP3 = take(128 * 4), bA0 = take(256 * 4), bA1 = take(128 * 4);
    const size_t bA2 = take( 32 * 4);

    const size_t act0_o  = take((size_t)MROWS * 32 * 2);   // bf16
    const size_t local_o = take((size_t)MROWS * 3 * 4);    // f32
    const size_t kdist_o = take((size_t)MROWS * 4);        // f32
    const size_t logit_o = take((size_t)MROWS * 4);        // f32
    const size_t bufA_o  = take((size_t)MROWS * 288 * 2);  // bf16 ping
    const size_t bufB_o  = take((size_t)MROWS * 288 * 2);  // bf16 pong

    bf16_t* act0  = (bf16_t*)(ws + act0_o);
    float*  locB  = (float*)(ws + local_o);
    float*  kdB   = (float*)(ws + kdist_o);
    float*  logit = (float*)(ws + logit_o);
    bf16_t* bufA  = (bf16_t*)(ws + bufA_o);
    bf16_t* bufB  = (bf16_t*)(ws + bufB_o);

    // --- 1) pack weights (BN folded) --------------------------------------
    auto pack = [&](BN L, int Ci, int Co, int Kc, int hasbn, size_t wo, size_t bo) {
        int total = (Co / 16) * Kc * 512;
        int grid  = (total + 255) / 256;
        hipLaunchKernelGGL(pack_weights, dim3(grid), dim3(256), 0, stream,
                           L.w, L.b, L.g, L.be, L.mu, L.va, Ci, Co, Kc, hasbn,
                           (bf16_t*)(ws + wo), (float*)(ws + bo));
    };
    pack(P0,   6,  64, 1, 1, wpP0, bP0);
    pack(P1,  64, 128, 2, 1, wpP1, bP1);
    pack(P2, 128, 128, 4, 1, wpP2, bP2);
    { BN L; L.w = P3w; L.b = P3b; L.g = L.be = L.mu = L.va = P3b; // unused when !bn
      pack(L, 128, 128, 4, 0, wpP3, bP3); }
    pack(A0, 263, 256, 9, 1, wpA0, bA0);
    pack(A1, 256, 128, 8, 1, wpA1, bA1);
    pack(A2, 128,  32, 4, 1, wpA2, bA2);

    // --- 2) KNN + geometry features ---------------------------------------
    hipLaunchKernelGGL(knn_kernel, dim3(NQ / 256, BATCH), dim3(256), 0, stream,
                       input_pcd, query_pts, act0, locB, kdB);

    // --- 3) GEMM chains ----------------------------------------------------
    // Each block covers 16 mtiles (8 waves x MB=2); grid.y = MTILES/16 = 640.
    auto gemm = [&](const bf16_t* in, int ldin, size_t wo, size_t bo,
                    bf16_t* out, int ldout, int kc, int N, int act) {
        const bf16_t* W = (const bf16_t*)(ws + wo);
        const float*  Bv = (const float*)(ws + bo);
        if (N >= 64) {   // NB = 4
            hipLaunchKernelGGL((gemm_bf16<4>), dim3(N / 64, MTILES / 16), dim3(256),
                               0, stream, in, ldin, W, Bv, out, ldout, kc, LRELU, act);
        } else {         // NB = 2 (N == 32)
            hipLaunchKernelGGL((gemm_bf16<2>), dim3(N / 32, MTILES / 16), dim3(256),
                               0, stream, in, ldin, W, Bv, out, ldout, kc, LRELU, act);
        }
    };

    // patch net
    gemm(act0, 32, wpP0, bP0, bufA,  64, 1,  64, 1);
    gemm(bufA, 64, wpP1, bP1, bufB, 128, 2, 128, 1);
    gemm(bufB, 128, wpP2, bP2, bufA, 128, 4, 128, 1);
    gemm(bufA, 128, wpP3, bP3, bufB, 128, 4, 128, 0);   // feat (no act)

    // --- 4) max-pool over K + build cv2 (288-wide) ------------------------
    hipLaunchKernelGGL(concat_maxpool, dim3(BATCH * NQ), dim3(160), 0, stream,
                       bufB, act0, kdB, bufA);

    // --- 5) attn net -------------------------------------------------------
    gemm(bufA, 288, wpA0, bA0, bufB, 256, 9, 256, 1);
    gemm(bufB, 256, wpA1, bA1, bufA, 128, 8, 128, 1);
    gemm(bufA, 128, wpA2, bA2, bufB,  32, 4,  32, 1);
    hipLaunchKernelGGL(attn_last, dim3(MROWS / 256), dim3(256), 0, stream,
                       bufB, A3w, A3b, logit);

    // --- 6) softmax + weighted sum + norm ---------------------------------
    hipLaunchKernelGGL(softmax_udf, dim3((BATCH * NQ + 255) / 256), dim3(256), 0, stream,
                       logit, locB, (float*)d_out);
}
